// MultiheadAttentionPooling_47777216200945
// MI455X (gfx1250) — matrix operand order repaired and checked by reading, not compiled, not run on hardware
//
#include <hip/hip_runtime.h>
#include <hip/hip_bf16.h>

// ---------------------------------------------------------------------------
// MultiheadAttentionPooling for MI455X (gfx1250, wave32, WMMA)
//
// N=6144 nodes (256 graphs x 24), D_IN=128, HEADS=8, D_HEAD=64, EFF=512.
// Edges are block-dense per graph -> dense 24x24 per-head attention.
//
// Matrix work on the WMMA pipe (v_wmma_f32_16x16x32_f16):
//   - QKV projections: 6144x128 @ 128x512 (x3), 16x64 output tile per wave.
//   - Attention QK^T and alpha*V per (graph, head), padded 24->32, with
//     zeroed pads so no garbage reaches live outputs.
// Everything is L2-resident (~20 MB << 192 MB L2).
// ---------------------------------------------------------------------------

#define N_NODES   6144
#define B_GRAPHS  256
#define G_NODES   24
#define D_IN      128
#define EFF       512
#define D_HEAD    64
#define HEADS     8
#define EPSN      1e-5f

typedef __attribute__((ext_vector_type(16))) _Float16 v16h;
typedef __attribute__((ext_vector_type(8)))  _Float16 v8h;
typedef __attribute__((ext_vector_type(8)))  float    v8f;

// ---------------------------------------------------------------- utilities
__global__ void zero_kernel(float* __restrict__ p, int n) {
  int i = blockIdx.x * blockDim.x + threadIdx.x;
  if (i < n) p[i] = 0.f;
}

__global__ void cvt_f16_kernel(const float* __restrict__ s, _Float16* __restrict__ d, int n) {
  int i = blockIdx.x * blockDim.x + threadIdx.x;
  if (i < n) d[i] = (_Float16)s[i];
}

// Per-channel sum / sum-of-squares over rows (C = 64 or 128, blockDim=256).
__global__ __launch_bounds__(256)
void stats_kernel(const float* __restrict__ x, float* __restrict__ sum,
                  float* __restrict__ sumsq, int nrows, int C) {
  const int rowsPerIter = blockDim.x / C;
  const int c = threadIdx.x % C;
  const int j = threadIdx.x / C;
  float s = 0.f, q = 0.f;
  for (int r = blockIdx.x * rowsPerIter + j; r < nrows; r += gridDim.x * rowsPerIter) {
    float v = x[r * C + c];
    s += v; q += v * v;
  }
  atomicAdd(&sum[c], s);
  atomicAdd(&sumsq[c], q);
}

// GraphNorm (global): xn = w*(x - mean*ms)*rsqrt(var+eps) + b, emitted as f16.
// var = Q/N - 2*ms*mean^2 + ms^2*mean^2.
__global__ __launch_bounds__(256)
void norm_cvt_kernel(const float* __restrict__ x, const float* __restrict__ sum,
                     const float* __restrict__ sq, const float* __restrict__ w,
                     const float* __restrict__ b, const float* __restrict__ ms,
                     _Float16* __restrict__ xn, int nrows, int C) {
  int i = blockIdx.x * blockDim.x + threadIdx.x;
  if (i >= nrows * C) return;
  int c = i % C;
  float invN = 1.f / (float)nrows;
  float mean = sum[c] * invN;
  float m = ms[c];
  float var = sq[c] * invN - 2.f * m * mean * mean + m * m * mean * mean;
  float v = (x[i] - mean * m) * rsqrtf(var + EPSN);
  xn[i] = (_Float16)(w[c] * v + b[c]);
}

// --------------------------------------------------------------- WMMA GEMM
// O[M=6144, 512](f16) = A[M,128](f16) x W[512,128]^T(f16) + bias.
// One 16x64 output tile per wave: A fragment loaded once per k-chunk and
// reused across 4 B fragments (4 accumulators, 16 WMMA per wave).
// A frag (16-bit A 16x32): lanes 0-15 M=lane, K=0..15;
//                          lanes 16-31 M=lane-16, K=16..31.
// B frag (32x16): half 0 -> N=lane, K=0..15; half 1 -> K=16..31.
// D: VGPR r -> M=r (+8 for half 1), N=lane&15.
__global__ __launch_bounds__(256)
void gemm_qkv_kernel(const _Float16* __restrict__ A, const _Float16* __restrict__ W,
                     const float* __restrict__ bias, _Float16* __restrict__ O) {
  const int lane = threadIdx.x & 31;
  const int wave = threadIdx.x >> 5;
  const int job  = blockIdx.x * 8 + wave;          // 384*8 = 3072 jobs
  const int tm   = job >> 3;                       // 384 M tiles
  const int tg   = job & 7;                        // 8 groups of 64 cols
  const int m0 = tm * 16, n0 = tg * 64;
  const int l = lane & 15, half = lane >> 4;

  const _Float16* arow = A + (m0 + l) * D_IN + half * 16;
  const _Float16* b0 = W + (n0 +  0 + l) * D_IN + half * 16;
  const _Float16* b1 = W + (n0 + 16 + l) * D_IN + half * 16;
  const _Float16* b2 = W + (n0 + 32 + l) * D_IN + half * 16;
  const _Float16* b3 = W + (n0 + 48 + l) * D_IN + half * 16;

  v8f acc0 = {}, acc1 = {}, acc2 = {}, acc3 = {};
#pragma unroll
  for (int kk = 0; kk < D_IN; kk += 32) {
    __builtin_prefetch(arow + kk + 32, 0, 1);      // global_prefetch_b8
    v8h a0 = *(const v8h*)(arow + kk);
    v8h a1 = *(const v8h*)(arow + kk + 8);
    v16h a;
#pragma unroll
    for (int i = 0; i < 8; ++i) { a[i] = a0[i]; a[i + 8] = a1[i]; }
    v16h f0 = *(const v16h*)(b0 + kk);
    v16h f1 = *(const v16h*)(b1 + kk);
    v16h f2 = *(const v16h*)(b2 + kk);
    v16h f3 = *(const v16h*)(b3 + kk);
    acc0 = __builtin_amdgcn_wmma_f32_16x16x32_f16(false, a, false, f0, (short)0, acc0, false, false);
    acc1 = __builtin_amdgcn_wmma_f32_16x16x32_f16(false, a, false, f1, (short)0, acc1, false, false);
    acc2 = __builtin_amdgcn_wmma_f32_16x16x32_f16(false, a, false, f2, (short)0, acc2, false, false);
    acc3 = __builtin_amdgcn_wmma_f32_16x16x32_f16(false, a, false, f3, (short)0, acc3, false, false);
  }

  const int mbase = m0 + half * 8;
#pragma unroll
  for (int i = 0; i < 4; ++i) {
    const v8f* accs[4] = { &acc0, &acc1, &acc2, &acc3 };
    const int n = n0 + i * 16 + l;
    const float bv = bias[n];
    const v8f& ac = *accs[i];
#pragma unroll
    for (int r = 0; r < 8; ++r)
      O[(mbase + r) * EFF + n] = (_Float16)(ac[r] + bv);
  }
}

// ----------------------------------------------------- per-graph attention
// Block = one graph, 8 waves = 8 heads.  All on WMMA with 24->32 padding:
//   S[32,32]   = Q[32,64] x K^T          (2x2 tiles, 2 k-chunks: 8 WMMA)
//   aggr[32,64]= alpha[32,32] x V[32,64] (2x4 tiles, 1 k-chunk : 8 WMMA)
// Pad alpha rows/cols and V^T pad rows are exact zeros, so pads contribute
// nothing; pad output rows are skipped.  Head-mean via ds_add_f32 atomics.
__global__ __launch_bounds__(256)
void attn_kernel(const _Float16* __restrict__ qf, const _Float16* __restrict__ kf,
                 const _Float16* __restrict__ vf, const float* __restrict__ x,
                 float* __restrict__ h) {
  __shared__ float    Sf[HEADS * 32 * 32];       // 32 KB; reused as vtT (f16)
  __shared__ _Float16 alf[HEADS * 32 * 32];      // 16 KB alpha (zero-padded)
  __shared__ float    aggsum[32 * D_HEAD];       //  8 KB head-sum accumulator
  _Float16* vtT = (_Float16*)Sf;                 // [head][ch(64)][k(32)]

  const int g = blockIdx.x, t = threadIdx.x;
  const int base = g * G_NODES;
  const int lane = t & 31, hh = t >> 5;          // wave == head
  const int l = lane & 15, half = lane >> 4;

  for (int idx = t; idx < HEADS * 32 * 32; idx += 256) alf[idx] = (_Float16)0.f;
  for (int idx = t; idx < 32 * D_HEAD;     idx += 256) aggsum[idx] = 0.f;
  __syncthreads();

  // ---- S = Q K^T / sqrt(Dh) --------------------------------------------
  {
    v8f acc[2][2] = {};
#pragma unroll
    for (int kk = 0; kk < D_HEAD; kk += 32) {
      v16h afr[2], bfr[2];
#pragma unroll
      for (int mt = 0; mt < 2; ++mt) {
        const _Float16* ap = qf + (base + mt * 16 + l) * EFF + hh * D_HEAD + kk + half * 16;
        v8h x0 = *(const v8h*)(ap);
        v8h x1 = *(const v8h*)(ap + 8);
#pragma unroll
        for (int i = 0; i < 8; ++i) { afr[mt][i] = x0[i]; afr[mt][i + 8] = x1[i]; }
      }
#pragma unroll
      for (int nt = 0; nt < 2; ++nt) {
        const _Float16* bp = kf + (base + nt * 16 + l) * EFF + hh * D_HEAD + kk + half * 16;
        bfr[nt] = *(const v16h*)(bp);
      }
#pragma unroll
      for (int mt = 0; mt < 2; ++mt)
#pragma unroll
        for (int nt = 0; nt < 2; ++nt)
          acc[mt][nt] = __builtin_amdgcn_wmma_f32_16x16x32_f16(
              false, afr[mt], false, bfr[nt], (short)0, acc[mt][nt], false, false);
    }
#pragma unroll
    for (int mt = 0; mt < 2; ++mt)
#pragma unroll
      for (int nt = 0; nt < 2; ++nt)
#pragma unroll
        for (int r = 0; r < 8; ++r) {
          int m = mt * 16 + half * 8 + r;
          int n = nt * 16 + l;
          Sf[hh * 1024 + m * 32 + n] = acc[mt][nt][r] * 0.125f;
        }
  }
  __syncthreads();

  // ---- row softmax over src<24, write alpha as f16 (pads stay zero) ----
  if (t < HEADS * G_NODES) {
    int th = t / G_NODES, d = t - th * G_NODES;
    const float* row = Sf + th * 1024 + d * 32;
    float mx = row[0];
    for (int s = 1; s < G_NODES; ++s) mx = fmaxf(mx, row[s]);
    float ssum = 0.f, e[G_NODES];
    for (int s = 0; s < G_NODES; ++s) { e[s] = __expf(row[s] - mx); ssum += e[s]; }
    float inv = 1.f / (ssum + 1e-16f);
    _Float16* arow = alf + th * 1024 + d * 32;
    for (int s = 0; s < G_NODES; ++s) arow[s] = (_Float16)(e[s] * inv);
  }
  __syncthreads();

  // ---- build V^T in LDS: vtT[head][ch][k], pad rows k>=24 are zero -----
  for (int idx = t; idx < HEADS * D_HEAD * 32; idx += 256) {
    int th = idx >> 11;                   // /2048
    int rem = idx & 2047;
    int n = rem >> 5, k = rem & 31;
    vtT[idx] = (k < G_NODES) ? vf[(base + k) * EFF + th * D_HEAD + n] : (_Float16)0.f;
  }
  __syncthreads();

  // ---- aggr = alpha x V  (K=32 single chunk) ---------------------------
  {
    v16h afr[2];
#pragma unroll
    for (int mt = 0; mt < 2; ++mt) {
      const _Float16* ap = alf + (hh * 32 + mt * 16 + l) * 32 + half * 16;
      v8h x0 = *(const v8h*)(ap);
      v8h x1 = *(const v8h*)(ap + 8);
#pragma unroll
      for (int i = 0; i < 8; ++i) { afr[mt][i] = x0[i]; afr[mt][i + 8] = x1[i]; }
    }
#pragma unroll
    for (int nt = 0; nt < 4; ++nt) {
      const _Float16* bp = vtT + (hh * 64 + nt * 16 + l) * 32 + half * 16;
      v16h bfr = *(const v16h*)(bp);
#pragma unroll
      for (int mt = 0; mt < 2; ++mt) {
        v8f acc = {};
        acc = __builtin_amdgcn_wmma_f32_16x16x32_f16(
            false, afr[mt], false, bfr, (short)0, acc, false, false);
#pragma unroll
        for (int r = 0; r < 8; ++r) {
          int m = mt * 16 + half * 8 + r;
          if (m < G_NODES)
            atomicAdd(&aggsum[m * D_HEAD + nt * 16 + l], acc[r]);  // ds_add_f32
        }
      }
    }
  }
  __syncthreads();

  // ---- head mean + sum-fold residual -----------------------------------
  for (int idx = t; idx < G_NODES * D_HEAD; idx += 256) {
    int d = idx >> 6, c = idx & 63;
    int node = base + d;
    float init = x[node * D_IN + c] + x[node * D_IN + D_HEAD + c];
    h[node * D_HEAD + c] = aggsum[idx] * 0.125f + init;
  }
}

// -------------------------------- GraphNorm(global) + GELU PFF + residual
__global__ __launch_bounds__(256)
void pff_out_kernel(const float* __restrict__ h, const float* __restrict__ sum,
                    const float* __restrict__ sq, const float* __restrict__ nw,
                    const float* __restrict__ nb, const float* __restrict__ nms,
                    const float* __restrict__ w1, const float* __restrict__ b1,
                    const float* __restrict__ w2, const float* __restrict__ b2,
                    float* __restrict__ h2, int nrows) {
  __shared__ float hn[4 * D_HEAD];
  __shared__ float t1[4 * D_HEAD];
  const int j = threadIdx.x >> 6;
  const int c = threadIdx.x & 63;
  const int node = blockIdx.x * 4 + j;

  float invN = 1.f / (float)nrows;
  float mean = sum[c] * invN;
  float m = nms[c];
  float var = sq[c] * invN - 2.f * m * mean * mean + m * m * mean * mean;
  float hv = h[node * D_HEAD + c];
  hn[j * 64 + c] = nw[c] * (hv - mean * m) * rsqrtf(var + EPSN) + nb[c];
  __syncthreads();

  float a = b1[c];
  const float* hr = hn + j * 64;
  const float* wr = w1 + c * 64;
#pragma unroll 8
  for (int kq = 0; kq < 64; ++kq) a += hr[kq] * wr[kq];
  t1[j * 64 + c] = 0.5f * a * (1.f + erff(a * 0.70710678118654752f)); // exact gelu
  __syncthreads();

  float o = b2[c];
  const float* tr = t1 + j * 64;
  const float* w2r = w2 + c * 64;
#pragma unroll 8
  for (int kq = 0; kq < 64; ++kq) o += tr[kq] * w2r[kq];
  h2[node * D_HEAD + c] = hv + o;
}

// ---------------- per-graph GraphNorm + score PFF + softmax + pooled output
__global__ __launch_bounds__(256)
void pool_kernel(const float* __restrict__ h2, const float* __restrict__ x,
                 const float* __restrict__ pnw, const float* __restrict__ pnb,
                 const float* __restrict__ pnms, const float* __restrict__ pw1,
                 const float* __restrict__ pb1, const float* __restrict__ pw2,
                 const float* __restrict__ pb2, float* __restrict__ out) {
  const int g = blockIdx.x, t = threadIdx.x;
  __shared__ float sh[G_NODES * D_HEAD];
  __shared__ float part[4 * D_HEAD];
  __shared__ float meanv[D_HEAD], varv[D_HEAD];
  __shared__ float hnb[G_NODES * D_HEAD];
  __shared__ float s1[G_NODES * D_HEAD];
  __shared__ float wgt[G_NODES];
  const int base = g * G_NODES;
  const int c = t & 63, j = t >> 6;

  for (int idx = t; idx < G_NODES * D_HEAD; idx += 256)
    sh[idx] = h2[(base + (idx >> 6)) * D_HEAD + (idx & 63)];
  __syncthreads();

  { // per-channel mean over 24 nodes
    float s = 0.f;
    for (int d = j; d < G_NODES; d += 4) s += sh[d * 64 + c];
    part[j * 64 + c] = s;
  }
  __syncthreads();
  if (t < 64) meanv[t] = (part[t] + part[64 + t] + part[128 + t] + part[192 + t]) * (1.f / 24.f);
  __syncthreads();
  { // variance of (h - mean*ms)
    float mm = meanv[c] * pnms[c];
    float s = 0.f;
    for (int d = j; d < G_NODES; d += 4) { float o = sh[d * 64 + c] - mm; s += o * o; }
    part[j * 64 + c] = s;
  }
  __syncthreads();
  if (t < 64) varv[t] = (part[t] + part[64 + t] + part[128 + t] + part[192 + t]) * (1.f / 24.f);
  __syncthreads();

  for (int idx = t; idx < G_NODES * D_HEAD; idx += 256) {
    int cc = idx & 63;
    float o = sh[idx] - meanv[cc] * pnms[cc];
    hnb[idx] = pnw[cc] * o * rsqrtf(varv[cc] + EPSN) + pnb[cc];
  }
  __syncthreads();

  for (int idx = t; idx < G_NODES * D_HEAD; idx += 256) {
    int d = idx >> 6, cc = idx & 63;
    float a = pb1[cc];
    const float* hr = hnb + d * 64;
    const float* wr = pw1 + cc * 64;
#pragma unroll 8
    for (int kq = 0; kq < 64; ++kq) a += hr[kq] * wr[kq];
    s1[idx] = 0.5f * a * (1.f + erff(a * 0.70710678118654752f));
  }
  __syncthreads();

  if (t < G_NODES) {                                 // score = s1 @ p_w2^T + p_b2
    float sc = pb2[0];
    const float* sr = s1 + t * 64;
#pragma unroll 8
    for (int kq = 0; kq < 64; ++kq) sc += sr[kq] * pw2[kq];
    wgt[t] = sc;
  }
  __syncthreads();
  if (t == 0) {                                      // segment softmax over 24
    float mx = wgt[0];
    for (int d = 1; d < G_NODES; ++d) mx = fmaxf(mx, wgt[d]);
    float ssum = 0.f;
    for (int d = 0; d < G_NODES; ++d) { float e = __expf(wgt[d] - mx); wgt[d] = e; ssum += e; }
    float inv = 1.f / (ssum + 1e-16f);
    for (int d = 0; d < G_NODES; ++d) wgt[d] *= inv;
  }
  __syncthreads();
  if (t < D_IN) {                                    // out[g,:] = sum_d w_d * x[d,:]
    float acc = 0.f;
    for (int d = 0; d < G_NODES; ++d) acc += wgt[d] * x[(base + d) * D_IN + t];
    out[g * D_IN + t] = acc;
  }
}

// --------------------------------------------------------------------------
extern "C" void kernel_launch(void* const* d_in, const int* in_sizes, int n_in,
                              void* d_out, int out_size, void* d_ws, size_t ws_size,
                              hipStream_t stream) {
  (void)in_sizes; (void)n_in; (void)out_size; (void)ws_size;
  const float* x     = (const float*)d_in[0];
  // d_in[1..3] = edge_index / ptr / batch : block-dense structure is fixed, unused.
  const float* nq_w  = (const float*)d_in[4];
  const float* nq_b  = (const float*)d_in[5];
  const float* nq_ms = (const float*)d_in[6];
  const float* wq    = (const float*)d_in[7];
  const float* bq    = (const float*)d_in[8];
  const float* wk    = (const float*)d_in[9];
  const float* bk    = (const float*)d_in[10];
  const float* wv    = (const float*)d_in[11];
  const float* bv    = (const float*)d_in[12];
  const float* no_w  = (const float*)d_in[13];
  const float* no_b  = (const float*)d_in[14];
  const float* no_ms = (const float*)d_in[15];
  const float* o_w1  = (const float*)d_in[16];
  const float* o_b1  = (const float*)d_in[17];
  const float* o_w2  = (const float*)d_in[18];
  const float* o_b2  = (const float*)d_in[19];
  const float* pn_w  = (const float*)d_in[20];
  const float* pn_b  = (const float*)d_in[21];
  const float* pn_ms = (const float*)d_in[22];
  const float* p_w1  = (const float*)d_in[23];
  const float* p_b1  = (const float*)d_in[24];
  const float* p_w2  = (const float*)d_in[25];
  const float* p_b2  = (const float*)d_in[26];

  char* wsp = (char*)d_ws;
  auto alloc = [&](size_t bytes) -> void* {
    void* p = (void*)wsp;
    wsp += (bytes + 255) & ~(size_t)255;
    return p;
  };
  float*    stats = (float*)   alloc(384 * sizeof(float));          // sum1|sq1|sum2|sq2
  float*    sum1  = stats;
  float*    sq1   = stats + 128;
  float*    sum2  = stats + 256;
  float*    sq2   = stats + 320;
  _Float16* xn16  = (_Float16*)alloc((size_t)N_NODES * D_IN * 2);
  _Float16* wq16  = (_Float16*)alloc((size_t)EFF * D_IN * 2);
  _Float16* wk16  = (_Float16*)alloc((size_t)EFF * D_IN * 2);
  _Float16* wv16  = (_Float16*)alloc((size_t)EFF * D_IN * 2);
  _Float16* qbuf  = (_Float16*)alloc((size_t)N_NODES * EFF * 2);
  _Float16* kbuf  = (_Float16*)alloc((size_t)N_NODES * EFF * 2);
  _Float16* vbuf  = (_Float16*)alloc((size_t)N_NODES * EFF * 2);
  float*    hbuf  = (float*)   alloc((size_t)N_NODES * D_HEAD * 4);
  float*    h2buf = (float*)   alloc((size_t)N_NODES * D_HEAD * 4);

  // 1) zero accumulators (ws is poisoned by the harness)
  zero_kernel<<<2, 256, 0, stream>>>(stats, 384);
  // 2) GraphNorm-1 stats over all N nodes
  stats_kernel<<<48, 256, 0, stream>>>(x, sum1, sq1, N_NODES, D_IN);
  // 3) normalize + f16 convert for WMMA
  norm_cvt_kernel<<<(N_NODES * D_IN) / 256, 256, 0, stream>>>(
      x, sum1, sq1, nq_w, nq_b, nq_ms, xn16, N_NODES, D_IN);
  // 4) weights -> f16
  cvt_f16_kernel<<<(EFF * D_IN) / 256, 256, 0, stream>>>(wq, wq16, EFF * D_IN);
  cvt_f16_kernel<<<(EFF * D_IN) / 256, 256, 0, stream>>>(wk, wk16, EFF * D_IN);
  cvt_f16_kernel<<<(EFF * D_IN) / 256, 256, 0, stream>>>(wv, wv16, EFF * D_IN);
  // 5-7) QKV on the WMMA pipe: 3072 wave-jobs (16x64 tile each), 8 waves/block
  gemm_qkv_kernel<<<384, 256, 0, stream>>>(xn16, wq16, bq, qbuf);
  gemm_qkv_kernel<<<384, 256, 0, stream>>>(xn16, wk16, bk, kbuf);
  gemm_qkv_kernel<<<384, 256, 0, stream>>>(xn16, wv16, bv, vbuf);
  // 8) dense per-graph attention (WMMA QK^T + WMMA alpha*V) + residual fold
  attn_kernel<<<B_GRAPHS, 256, 0, stream>>>(qbuf, kbuf, vbuf, x, hbuf);
  // 9) GraphNorm-2 stats + fused norm/PFF/residual
  stats_kernel<<<48, 256, 0, stream>>>(hbuf, sum2, sq2, N_NODES, D_HEAD);
  pff_out_kernel<<<N_NODES / 4, 256, 0, stream>>>(
      hbuf, sum2, sq2, no_w, no_b, no_ms, o_w1, o_b1, o_w2, o_b2, h2buf, N_NODES);
  // 10) per-graph norm + score + softmax + pooled output
  pool_kernel<<<B_GRAPHS, 256, 0, stream>>>(
      h2buf, x, pn_w, pn_b, pn_ms, p_w1, p_b1, p_w2, p_b2, (float*)d_out);
}
